// ScannedRNN_41429254537265
// MI455X (gfx1250) — matrix operand order repaired and checked by reading, not compiled, not run on hardware
//
#include <hip/hip_runtime.h>
#include <hip/hip_bf16.h>

// ---------------------------------------------------------------------------
// LSTM scan with reset masking, tuned for MI455X (gfx1250):
//  - bf16 WMMA (v_wmma_f32_16x16x32_bf16) for x@Wi + h@Wh fused as one
//    K=1536 GEMM against a pre-transposed combined weight (12MB bf16,
//    L2-resident across all 256 sequential steps).
//  - Reset masking of h is pre-applied once per step by the pointwise kernel
//    (using resets[t+1]), so the GEMM inner loop is pure load+WMMA.
//  - fp32 accumulation + fp32 pointwise gate math.
// ---------------------------------------------------------------------------

typedef __attribute__((ext_vector_type(16))) __bf16 v16bf;
typedef __attribute__((ext_vector_type(8)))  __bf16 v8bf;
typedef __attribute__((ext_vector_type(8)))  float  v8f;

#define T_STEPS 256
#define BATCH   256
#define OBSD    512
#define FEAT    1024
#define N4F     4096
#define KTOT    1536   // OBSD + FEAT

// ---------------------------------------------------------------------------
// Prep: tiled transpose fp32 [Ks][4096] -> bf16 Wt[4096][1536] (+kOffset)
// ---------------------------------------------------------------------------
__global__ void transpose_f32_to_bf16(const float* __restrict__ src,
                                      __bf16* __restrict__ dst,
                                      int kOffset) {
  __shared__ float tile[32][33];
  const int n0 = blockIdx.x * 32;
  const int k0 = blockIdx.y * 32;
  const int tx = threadIdx.x;   // 0..31
  const int ty = threadIdx.y;   // 0..7
#pragma unroll
  for (int r = 0; r < 4; ++r)
    tile[ty + 8 * r][tx] = src[(size_t)(k0 + ty + 8 * r) * N4F + n0 + tx];
  __syncthreads();
#pragma unroll
  for (int r = 0; r < 4; ++r)
    dst[(size_t)(n0 + ty + 8 * r) * KTOT + kOffset + k0 + tx] =
        (__bf16)tile[tx][ty + 8 * r];
}

// ---------------------------------------------------------------------------
// Prep: flat fp32 -> bf16 (obs), 4 elements/thread
// ---------------------------------------------------------------------------
__global__ void f32_to_bf16_vec4(const float* __restrict__ src,
                                 __bf16* __restrict__ dst, int n4) {
  const int id = blockIdx.x * blockDim.x + threadIdx.x;
  if (id < n4) {
    float4 v = ((const float4*)src)[id];
    dst[4 * id + 0] = (__bf16)v.x;
    dst[4 * id + 1] = (__bf16)v.y;
    dst[4 * id + 2] = (__bf16)v.z;
    dst[4 * id + 3] = (__bf16)v.w;
  }
}

// ---------------------------------------------------------------------------
// Prep: carry init; h0 is pre-masked with resets[0] for step 0's GEMM.
// ---------------------------------------------------------------------------
__global__ void init_carry(const float* __restrict__ h0,
                           const float* __restrict__ c0,
                           const unsigned char* __restrict__ rt0,
                           __bf16* __restrict__ hmask,
                           float* __restrict__ c, int n) {
  const int id = blockIdx.x * blockDim.x + threadIdx.x;
  if (id < n) {
    hmask[id] = rt0[id >> 10] ? (__bf16)0.0f : (__bf16)h0[id];
    c[id]     = c0[id];
  }
}

// ---------------------------------------------------------------------------
// WMMA helper: run one K-span. A fragment (16x32 bf16): lane m holds K
// {kb..kb+7, kb+16..kb+23}, kb = 8*(lane>=16) -> two 16B loads. B fragment
// (32x16 bf16): lane n holds the same K set as A -> two 16B loads from the
// pre-transposed Wt row.
// ---------------------------------------------------------------------------
__device__ __forceinline__ void mma_span(v8f (&acc)[2][2],
                                         const __bf16* __restrict__ a0,
                                         const __bf16* __restrict__ a1,
                                         const __bf16* __restrict__ w0,
                                         const __bf16* __restrict__ w1,
                                         int kLen) {
  for (int k0 = 0; k0 < kLen; k0 += 32) {
    v16bf af[2];
    {
      v8bf lo  = *(const v8bf*)(a0 + k0);
      v8bf ahi = *(const v8bf*)(a0 + k0 + 16);
      af[0] = __builtin_shufflevector(lo, ahi, 0, 1, 2, 3, 4, 5, 6, 7, 8, 9,
                                      10, 11, 12, 13, 14, 15);
    }
    {
      v8bf lo  = *(const v8bf*)(a1 + k0);
      v8bf ahi = *(const v8bf*)(a1 + k0 + 16);
      af[1] = __builtin_shufflevector(lo, ahi, 0, 1, 2, 3, 4, 5, 6, 7, 8, 9,
                                      10, 11, 12, 13, 14, 15);
    }
    {
      v16bf bf = __builtin_shufflevector(*(const v8bf*)(w0 + k0), *(const v8bf*)(w0 + k0 + 16), 0, 1, 2, 3, 4, 5, 6, 7, 8, 9, 10, 11, 12, 13, 14, 15);
      acc[0][0] = __builtin_amdgcn_wmma_f32_16x16x32_bf16(
          false, af[0], false, bf, (short)0, acc[0][0], false, false);
      acc[1][0] = __builtin_amdgcn_wmma_f32_16x16x32_bf16(
          false, af[1], false, bf, (short)0, acc[1][0], false, false);
    }
    {
      v16bf bf = __builtin_shufflevector(*(const v8bf*)(w1 + k0), *(const v8bf*)(w1 + k0 + 16), 0, 1, 2, 3, 4, 5, 6, 7, 8, 9, 10, 11, 12, 13, 14, 15);
      acc[0][1] = __builtin_amdgcn_wmma_f32_16x16x32_bf16(
          false, af[0], false, bf, (short)0, acc[0][1], false, false);
      acc[1][1] = __builtin_amdgcn_wmma_f32_16x16x32_bf16(
          false, af[1], false, bf, (short)0, acc[1][1], false, false);
    }
  }
}

// ---------------------------------------------------------------------------
// Per-step GEMM:  gates[256][4096] = [x_t | h_masked] (256x1536) @ Wt^T + b
// Block: 256 threads = 8 waves arranged 2(M) x 4(N); wave tile 32x32
// (2 M-tiles x 2 N-tiles of 16x16). Block tile 64x128 -> grid (4, 32).
// EXEC is all-1s everywhere (no divergent control flow), as WMMA requires.
// ---------------------------------------------------------------------------
__global__ __launch_bounds__(256) void lstm_gemm_step(
    const __bf16* __restrict__ xt,     // [256][512] bf16, this timestep
    const __bf16* __restrict__ hm,     // [256][1024] bf16, pre-masked h
    const __bf16* __restrict__ Wt,     // [4096][1536] bf16 (transposed)
    const float* __restrict__ bias,    // [4096]
    float* __restrict__ gates)         // [256][4096]
{
  const int lane  = threadIdx.x & 31;
  const int wave  = threadIdx.x >> 5;
  const int waveM = wave & 1;
  const int waveN = wave >> 1;
  const int mBlk  = blockIdx.x * 64 + waveM * 32;
  const int nBlk  = blockIdx.y * 128 + waveN * 32;

  const int lm   = lane & 15;
  const int hi16 = lane >> 4;       // 0 or 1
  const int kA   = hi16 << 3;       // A frag: +8 for upper half-wave
  const int kB   = hi16 << 3;       // B frag: +8 for upper half-wave

  v8f zero = {};
  v8f acc[2][2];
#pragma unroll
  for (int i = 0; i < 2; ++i)
#pragma unroll
    for (int j = 0; j < 2; ++j) acc[i][j] = zero;

  // Per-lane row base pointers (constant across the K loops).
  const __bf16* w0 = Wt + (size_t)(nBlk + lm) * KTOT + kB;
  const __bf16* w1 = Wt + (size_t)(nBlk + 16 + lm) * KTOT + kB;

  // x region: K = 0..511
  mma_span(acc,
           xt + (size_t)(mBlk + lm) * OBSD + kA,
           xt + (size_t)(mBlk + 16 + lm) * OBSD + kA,
           w0, w1, OBSD);
  // h region: K = 512..1535 (h already reset-masked)
  mma_span(acc,
           hm + (size_t)(mBlk + lm) * FEAT + kA,
           hm + (size_t)(mBlk + 16 + lm) * FEAT + kA,
           w0 + OBSD, w1 + OBSD, FEAT);

  // Epilogue: C/D 16x16 f32 layout: VGPR v -> row v + 8*(lane>=16), col lane&15
#pragma unroll
  for (int j = 0; j < 2; ++j) {
    const int col  = nBlk + 16 * j + lm;
    const float bv = bias[col];
#pragma unroll
    for (int i = 0; i < 2; ++i) {
      const int rbase = mBlk + 16 * i + (hi16 << 3);
#pragma unroll
      for (int v = 0; v < 8; ++v)
        gates[(size_t)(rbase + v) * N4F + col] = acc[i][j][v] + bv;
    }
  }
}

// ---------------------------------------------------------------------------
// Per-step pointwise: gate activations, carry update, outputs. Also emits the
// pre-masked bf16 h for the NEXT step's GEMM (mask = resets[t+1]).
// ---------------------------------------------------------------------------
__device__ __forceinline__ float sigm(float x) {
  return 1.0f / (1.0f + __expf(-x));
}

__global__ void lstm_pointwise(const float* __restrict__ gates,  // [256][4096]
                               const unsigned char* __restrict__ rtCur,
                               const unsigned char* __restrict__ rtNext,
                               float* __restrict__ c,            // [256][1024]
                               __bf16* __restrict__ hmask,       // [256][1024]
                               float* __restrict__ emb,          // d_out slice
                               float* __restrict__ hT,
                               float* __restrict__ cT,
                               int writeFinal) {
  const int id = blockIdx.x * blockDim.x + threadIdx.x;  // 0..262143
  const int b  = id >> 10;
  const int f  = id & 1023;
  const float* g0 = gates + (size_t)b * N4F;
  const float gi = g0[f];
  const float gf = g0[FEAT + f];
  const float gg = g0[2 * FEAT + f];
  const float go = g0[3 * FEAT + f];
  const float cp = rtCur[b] ? 0.0f : c[id];
  const float nc = sigm(gf) * cp + sigm(gi) * tanhf(gg);
  const float nh = sigm(go) * tanhf(nc);
  c[id]     = nc;
  emb[id]   = nh;
  hmask[id] = rtNext[b] ? (__bf16)0.0f : (__bf16)nh;
  if (writeFinal) {
    hT[id] = nh;
    cT[id] = nc;
  }
}

// ---------------------------------------------------------------------------
// Host launcher
// ---------------------------------------------------------------------------
extern "C" void kernel_launch(void* const* d_in, const int* in_sizes, int n_in,
                              void* d_out, int out_size, void* d_ws,
                              size_t ws_size, hipStream_t stream) {
  (void)in_sizes; (void)n_in; (void)out_size; (void)ws_size;

  const float*         obs    = (const float*)d_in[0];          // [T][B][OBS]
  const unsigned char* resets = (const unsigned char*)d_in[1];  // [T][B] bool
  const float*         h0     = (const float*)d_in[2];          // [B][F]
  const float*         c0     = (const float*)d_in[3];          // [B][F]
  const float*         Wi     = (const float*)d_in[4];          // [OBS][4F]
  const float*         Wh     = (const float*)d_in[5];          // [F][4F]
  const float*         bias   = (const float*)d_in[6];          // [4F]

  char* ws = (char*)d_ws;
  size_t off = 0;
  __bf16* Wt    = (__bf16*)(ws + off); off += (size_t)N4F * KTOT * 2;            // 12 MB
  __bf16* obsb  = (__bf16*)(ws + off); off += (size_t)T_STEPS * BATCH * OBSD * 2;// 64 MB
  float*  gates = (float*)(ws + off);  off += (size_t)BATCH * N4F * 4;           // 4 MB
  float*  c     = (float*)(ws + off);  off += (size_t)BATCH * FEAT * 4;          // 1 MB
  __bf16* hmask = (__bf16*)(ws + off);                                           // 0.5 MB

  float* out = (float*)d_out;
  float* hT  = out;
  float* cT  = out + (size_t)BATCH * FEAT;
  float* emb = out + (size_t)2 * BATCH * FEAT;

  // Prep: weights (transposed, bf16, combined [Wi;Wh] along K), obs, carry.
  {
    dim3 blk(32, 8);
    transpose_f32_to_bf16<<<dim3(N4F / 32, OBSD / 32), blk, 0, stream>>>(Wi, Wt, 0);
    transpose_f32_to_bf16<<<dim3(N4F / 32, FEAT / 32), blk, 0, stream>>>(Wh, Wt, OBSD);
  }
  {
    const int n4 = T_STEPS * BATCH * OBSD / 4;
    f32_to_bf16_vec4<<<(n4 + 255) / 256, 256, 0, stream>>>(obs, obsb, n4);
  }
  {
    const int n = BATCH * FEAT;
    init_carry<<<(n + 255) / 256, 256, 0, stream>>>(h0, c0, resets, hmask, c, n);
  }

  // Sequential scan over T (stream order provides the dependence chain).
  for (int t = 0; t < T_STEPS; ++t) {
    const int tn = (t + 1 < T_STEPS) ? (t + 1) : t;  // resets[t+1] (clamped)
    lstm_gemm_step<<<dim3(BATCH / 64, N4F / 128), 256, 0, stream>>>(
        obsb + (size_t)t * BATCH * OBSD, hmask, Wt, bias, gates);
    lstm_pointwise<<<(BATCH * FEAT) / 256, 256, 0, stream>>>(
        gates, resets + (size_t)t * BATCH, resets + (size_t)tn * BATCH, c,
        hmask, emb + (size_t)t * BATCH * FEAT, hT, cT, t == T_STEPS - 1);
  }
}
